// Attention_7842610282627
// MI455X (gfx1250) — matrix operand order repaired and checked by reading, not compiled, not run on hardware
//
#include <hip/hip_runtime.h>
#include <hip/hip_bf16.h>

#define NN 8192
#define DD 256
#define ROWS 32                      // query rows per block
// softmax scale folded with log2(e): scores/sqrt(256) * log2(e)
#define C2 (0.0625f * 1.44269504088896f)

// base-2 exponential -> v_exp_f32
#define EXP2F(x) __builtin_amdgcn_exp2f(x)

typedef __attribute__((ext_vector_type(16))) __bf16 bf16x16;
typedef __attribute__((ext_vector_type(8)))  float  floatx8;

union BFrag {
    bf16x16 v;
    uint4 u[2];
    __bf16 e[16];
};

// A-matrix fragment (16x32 bf16), row-major source with row stride `stride` (elements).
// Lane holds row m; element i -> K = kbase + (i>>3)*16 + h*8 + (i&7).
__device__ __forceinline__ bf16x16 load_a_frag(const __bf16* src, int stride,
                                               int m, int h, int kbase) {
    BFrag f;
    f.u[0] = *(const uint4*)(src + m * stride + kbase + h * 8);
    f.u[1] = *(const uint4*)(src + m * stride + kbase + 16 + h * 8);
    return f.v;
}

// B-matrix fragment (32x16 bf16) from row-major bf16 where B[k][n] = M[row_n][k].
// Lane holds column n = row_n; kstart = kbase + h*16; element i -> K = kbase + (i>>3)*16 + h*8 + (i&7).
__device__ __forceinline__ bf16x16 load_b_frag(const __bf16* g, size_t stride,
                                               int row, int kstart) {
    const __bf16* p = g + (size_t)row * stride + kstart - ((kstart & 16) >> 1);
    BFrag f;
    f.u[0] = *(const uint4*)(p);
    f.u[1] = *(const uint4*)(p + 16);
    return f.v;
}

#define WMMA_BF16(a, b, c) \
    __builtin_amdgcn_wmma_f32_16x16x32_bf16(false, (a), false, (b), (short)0, (c), false, false)

// CDNA5 async memory->LDS copy, 16 bytes per lane, tracked by ASYNCcnt.
__device__ __forceinline__ void async_copy_b128(unsigned lds_addr, unsigned long long gaddr) {
    asm volatile("global_load_async_to_lds_b128 %0, %1, off"
                 :: "v"(lds_addr), "v"(gaddr) : "memory");
}
__device__ __forceinline__ void wait_asynccnt0() {
    asm volatile("s_wait_asynccnt 0x0" ::: "memory");
}

// ---------------------------------------------------------------------------
// Kernel 0: one-shot fp32 -> bf16 conversion of the three weight matrices.
// ---------------------------------------------------------------------------
__global__ __launch_bounds__(256) void convert_w_kernel(
    const float* __restrict__ Wq, const float* __restrict__ Wk, const float* __restrict__ Wv,
    __bf16* __restrict__ Wqb, __bf16* __restrict__ Wkb, __bf16* __restrict__ Wvb) {
    const int mat = blockIdx.y;
    const float* src = (mat == 0) ? Wq : (mat == 1) ? Wk : Wv;
    __bf16* dst = (mat == 0) ? Wqb : (mat == 1) ? Wkb : Wvb;
    const int idx = blockIdx.x * 256 + threadIdx.x;   // one float4 per thread
    float4 f = ((const float4*)src)[idx];
    union { unsigned long long u; __bf16 e[4]; } o;
    o.e[0] = (__bf16)f.x;
    o.e[1] = (__bf16)f.y;
    o.e[2] = (__bf16)f.z;
    o.e[3] = (__bf16)f.w;
    ((unsigned long long*)dst)[idx] = o.u;
}

// ---------------------------------------------------------------------------
// Kernel 1: Q = xWq^T, K = xWk^T, V = xWv^T. Q,K row-major bf16; V transposed.
// One block = 16 rows of x; 8 waves cover 48 16x16 output tiles (6 each).
// ---------------------------------------------------------------------------
__global__ __launch_bounds__(256) void qkv_proj_kernel(
    const float* __restrict__ x,
    const __bf16* __restrict__ Wqb, const __bf16* __restrict__ Wkb,
    const __bf16* __restrict__ Wvb,
    __bf16* __restrict__ Qb, __bf16* __restrict__ Kb, __bf16* __restrict__ Vt) {
    __shared__ __bf16 xs[16 * DD];   // 8 KB: x tile as bf16

    const int tid = threadIdx.x;
    const int w = tid >> 5;
    const int lane = tid & 31;
    const int nl = lane & 15;
    const int h = lane >> 4;
    const int rowbase = blockIdx.x * 16;

    // Stage x[rowbase:rowbase+16][0:256] -> bf16 in LDS (16 elems/thread).
    {
        const int base = tid * 16;
        const int r = base >> 8, c = base & 255;
        const float4* xf = (const float4*)(x + (size_t)(rowbase + r) * DD + c);
#pragma unroll
        for (int q = 0; q < 4; ++q) {
            float4 d = xf[q];
            xs[base + 4 * q + 0] = (__bf16)d.x;
            xs[base + 4 * q + 1] = (__bf16)d.y;
            xs[base + 4 * q + 2] = (__bf16)d.z;
            xs[base + 4 * q + 3] = (__bf16)d.w;
        }
    }
    __syncthreads();

    for (int t = 0; t < 6; ++t) {
        const int tile = w * 6 + t;        // 0..47
        const int mat = tile >> 4;         // 0=Q 1=K 2=V
        const int colbase = (tile & 15) * 16;
        const __bf16* Wm = (mat == 0) ? Wqb : (mat == 1) ? Wkb : Wvb;

        floatx8 acc = {};
#pragma unroll
        for (int ks = 0; ks < 8; ++ks) {
            bf16x16 a = load_a_frag(xs, DD, nl, h, ks * 32);
            bf16x16 b = load_b_frag(Wm, DD, colbase + nl, ks * 32 + h * 16);
            acc = WMMA_BF16(a, b, acc);
        }

        if (mat < 2) {
            __bf16* dst = (mat == 0) ? Qb : Kb;
#pragma unroll
            for (int v = 0; v < 8; ++v) {
                int m_ = v + 8 * h;
                dst[(size_t)(rowbase + m_) * DD + colbase + nl] = (__bf16)acc[v];
            }
        } else {
#pragma unroll
            for (int v = 0; v < 8; ++v) {
                int m_ = v + 8 * h;
                Vt[(size_t)(colbase + nl) * NN + rowbase + m_] = (__bf16)acc[v];
            }
        }
    }
}

// ---------------------------------------------------------------------------
// Kernel 2: fused scores + softmax + attn output + cntx = attn @ V.
// One block = 32 query rows (2 row-tiles shared by every wave, so each loaded
// K/V/P fragment feeds two WMMAs).  Phase 1: online (max, sumexp2) over all
// keys (single TRANS op per element).  Phase 2: recompute S, write normalized
// attn (fp32), accumulate cntx via WMMA.
// ---------------------------------------------------------------------------
__global__ __launch_bounds__(256) void attention_kernel(
    const __bf16* __restrict__ Qb, const __bf16* __restrict__ Kb,
    const __bf16* __restrict__ Vt,
    float* __restrict__ cntx, float* __restrict__ attn) {
    __shared__ __bf16 qs[ROWS * DD];    // 16 KB  Q rows bf16
    __shared__ __bf16 ps[ROWS * 128];   // 8 KB   P tile bf16
    __shared__ float red[8][ROWS][2];   // per-wave (max2, sum) per row
    __shared__ float rowm[ROWS];
    __shared__ float rowinv[ROWS];

    const int tid = threadIdx.x;
    const int w = tid >> 5;
    const int lane = tid & 31;
    const int nl = lane & 15;
    const int h = lane >> 4;
    const int rowbase = blockIdx.x * ROWS;

    // Stage Q rows (already bf16) into LDS with CDNA5 async DMA:
    // 16 KB = 1024 x b128, 4 per thread, tracked by ASYNCcnt.
    {
        unsigned long long g = (unsigned long long)(const char*)(Qb + (size_t)rowbase * DD);
        unsigned l = (unsigned)(size_t)&qs[0];
#pragma unroll
        for (int q = 0; q < 4; ++q) {
            unsigned off = (unsigned)(tid + q * 256) * 16u;
            async_copy_b128(l + off, g + off);
        }
        wait_asynccnt0();
    }
    __syncthreads();

    // ---------------- Phase 1: row max & sum(exp2) over all keys -------------
    float mx[2][8], sm[2][8];
#pragma unroll
    for (int rt = 0; rt < 2; ++rt)
#pragma unroll
        for (int v = 0; v < 8; ++v) { mx[rt][v] = -INFINITY; sm[rt][v] = 0.0f; }

    for (int jt = w; jt < NN / 16; jt += 8) {
        const int jbase = jt * 16;
        floatx8 acc0 = {};
        floatx8 acc1 = {};
#pragma unroll
        for (int ks = 0; ks < 8; ++ks) {
            bf16x16 b = load_b_frag(Kb, DD, jbase + nl, ks * 32 + h * 16);
            bf16x16 a0 = load_a_frag(qs, DD, nl, h, ks * 32);
            bf16x16 a1 = load_a_frag(qs, DD, 16 + nl, h, ks * 32);
            acc0 = WMMA_BF16(a0, b, acc0);
            acc1 = WMMA_BF16(a1, b, acc1);
        }
#pragma unroll
        for (int rt = 0; rt < 2; ++rt) {
#pragma unroll
            for (int v = 0; v < 8; ++v) {
                // online update with one exp2: one of the two exponents is 0.
                float s2 = (rt ? acc1[v] : acc0[v]) * C2;
                float d = s2 - mx[rt][v];
                float t = EXP2F(0.0f - fabsf(d));
                bool gt = d > 0.0f;
                sm[rt][v] = gt ? fmaf(sm[rt][v], t, 1.0f) : (sm[rt][v] + t);
                mx[rt][v] = gt ? s2 : mx[rt][v];
            }
        }
    }
    // Butterfly reduce across the 16 lanes sharing each row (width-16 halves).
#pragma unroll
    for (int rt = 0; rt < 2; ++rt) {
#pragma unroll
        for (int v = 0; v < 8; ++v) {
#pragma unroll
            for (int off = 1; off < 16; off <<= 1) {
                float m2 = __shfl_xor(mx[rt][v], off, 32);
                float l2 = __shfl_xor(sm[rt][v], off, 32);
                float d = m2 - mx[rt][v];
                float t = EXP2F(0.0f - fabsf(d));
                bool gt = d > 0.0f;
                sm[rt][v] = gt ? fmaf(sm[rt][v], t, l2) : fmaf(l2, t, sm[rt][v]);
                mx[rt][v] = gt ? m2 : mx[rt][v];
            }
        }
    }
    if (nl == 0) {
#pragma unroll
        for (int rt = 0; rt < 2; ++rt)
#pragma unroll
            for (int v = 0; v < 8; ++v) {
                red[w][rt * 16 + v + 8 * h][0] = mx[rt][v];
                red[w][rt * 16 + v + 8 * h][1] = sm[rt][v];
            }
    }
    __syncthreads();
    if (tid < ROWS) {
        float m = red[0][tid][0];
        float l = red[0][tid][1];
        for (int wv = 1; wv < 8; ++wv) {
            float m2 = red[wv][tid][0], l2 = red[wv][tid][1];
            float d = m2 - m;
            float t = EXP2F(0.0f - fabsf(d));
            bool gt = d > 0.0f;
            l = gt ? fmaf(l, t, l2) : fmaf(l2, t, l);
            m = gt ? m2 : m;
        }
        rowm[tid] = m;
        rowinv[tid] = 1.0f / l;
    }
    __syncthreads();

    // ---------------- Phase 2: attn output + cntx accumulation ---------------
    floatx8 cacc[2][2] = {};             // [row-tile][col-tile]
    float rmv[2][8], riv[2][8];
#pragma unroll
    for (int rt = 0; rt < 2; ++rt)
#pragma unroll
        for (int v = 0; v < 8; ++v) {
            rmv[rt][v] = rowm[rt * 16 + v + 8 * h];
            riv[rt][v] = rowinv[rt * 16 + v + 8 * h];
        }

    for (int jblk = 0; jblk < NN / 128; ++jblk) {
        const int jbase0 = jblk * 128;
        const int jbase = jbase0 + w * 16;

        // S tiles (both row-tiles) for this wave's 16 keys.
        floatx8 acc0 = {};
        floatx8 acc1 = {};
#pragma unroll
        for (int ks = 0; ks < 8; ++ks) {
            bf16x16 b = load_b_frag(Kb, DD, jbase + nl, ks * 32 + h * 16);
            bf16x16 a0 = load_a_frag(qs, DD, nl, h, ks * 32);
            bf16x16 a1 = load_a_frag(qs, DD, 16 + nl, h, ks * 32);
            acc0 = WMMA_BF16(a0, b, acc0);
            acc1 = WMMA_BF16(a1, b, acc1);
        }
        // Normalized probabilities -> global attn (fp32) + LDS P tile (bf16).
#pragma unroll
        for (int rt = 0; rt < 2; ++rt) {
#pragma unroll
            for (int v = 0; v < 8; ++v) {
                int m_ = rt * 16 + v + 8 * h;
                float s = rt ? acc1[v] : acc0[v];
                float p = EXP2F(fmaf(s, C2, -rmv[rt][v])) * riv[rt][v];
                attn[(size_t)(rowbase + m_) * NN + jbase + nl] = p;
                ps[m_ * 128 + w * 16 + nl] = (__bf16)p;
            }
        }
        __syncthreads();

        // cntx += P[32x128] @ V[128 x 32-col slice owned by this wave].
#pragma unroll
        for (int ks = 0; ks < 4; ++ks) {
            bf16x16 a0 = load_a_frag(ps, 128, nl, h, ks * 32);
            bf16x16 a1 = load_a_frag(ps, 128, 16 + nl, h, ks * 32);
            bf16x16 b0 = load_b_frag(Vt, NN, w * 32 + nl, jbase0 + ks * 32 + h * 16);
            bf16x16 b1 = load_b_frag(Vt, NN, w * 32 + 16 + nl, jbase0 + ks * 32 + h * 16);
            cacc[0][0] = WMMA_BF16(a0, b0, cacc[0][0]);
            cacc[0][1] = WMMA_BF16(a0, b1, cacc[0][1]);
            cacc[1][0] = WMMA_BF16(a1, b0, cacc[1][0]);
            cacc[1][1] = WMMA_BF16(a1, b1, cacc[1][1]);
        }
        __syncthreads();
    }

#pragma unroll
    for (int rt = 0; rt < 2; ++rt)
#pragma unroll
        for (int v = 0; v < 8; ++v) {
            int m_ = rt * 16 + v + 8 * h;
            cntx[(size_t)(rowbase + m_) * DD + w * 32 + nl] = cacc[rt][0][v];
            cntx[(size_t)(rowbase + m_) * DD + w * 32 + 16 + nl] = cacc[rt][1][v];
        }
}

// ---------------------------------------------------------------------------
extern "C" void kernel_launch(void* const* d_in, const int* in_sizes, int n_in,
                              void* d_out, int out_size, void* d_ws, size_t ws_size,
                              hipStream_t stream) {
    const float* x  = (const float*)d_in[0];
    const float* Wq = (const float*)d_in[1];
    const float* Wk = (const float*)d_in[2];
    const float* Wv = (const float*)d_in[3];

    // Workspace: Qb, Kb row-major bf16 [N,D]; Vt transposed bf16 [D,N];
    // Wqb/Wkb/Wvb bf16 [D,D].  ~12.8 MB total.
    __bf16* Qb  = (__bf16*)d_ws;
    __bf16* Kb  = Qb + (size_t)NN * DD;
    __bf16* Vt  = Kb + (size_t)NN * DD;
    __bf16* Wqb = Vt + (size_t)NN * DD;
    __bf16* Wkb = Wqb + (size_t)DD * DD;
    __bf16* Wvb = Wkb + (size_t)DD * DD;

    float* cntx = (float*)d_out;                  // [N, D]
    float* attn = cntx + (size_t)NN * DD;         // [N, N]

    convert_w_kernel<<<dim3(DD * DD / 4 / 256, 3), 256, 0, stream>>>(
        Wq, Wk, Wv, Wqb, Wkb, Wvb);
    qkv_proj_kernel<<<dim3(NN / 16), 256, 0, stream>>>(x, Wqb, Wkb, Wvb, Qb, Kb, Vt);
    attention_kernel<<<dim3(NN / ROWS), 256, 0, stream>>>(Qb, Kb, Vt, cntx, attn);
}
